// GraphAttentionLayer_37606733644546
// MI455X (gfx1250) — matrix operand order repaired and checked by reading, not compiled, not run on hardware
//
#include <hip/hip_runtime.h>
#include <hip/hip_bf16.h>

// Problem constants from the reference
#define BATCH 8
#define NN    2048
#define INF   512
#define OUTF  128
// ratio = max(0.4 / 2^0, 0.1) = 0.4f

typedef __attribute__((ext_vector_type(16))) __bf16 v16bf;
typedef __attribute__((ext_vector_type(8)))  float  v8f;
typedef __attribute__((ext_vector_type(4)))  float  v4f;
typedef __attribute__((ext_vector_type(4)))  int    v4i;

// ---------------------------------------------------------------------------
// Kernel 1: wa1[i] = sum_o W[o,i]*a1[o], wa2[i] = sum_o W[o,i]*a2[o]
// ---------------------------------------------------------------------------
__global__ void gat_prep(const float* __restrict__ W, const float* __restrict__ a1,
                         const float* __restrict__ a2,
                         float* __restrict__ wa1, float* __restrict__ wa2) {
    int i = blockIdx.x * blockDim.x + threadIdx.x;
    if (i >= INF) return;
    float s1 = 0.f, s2 = 0.f;
    #pragma unroll 4
    for (int o = 0; o < OUTF; ++o) {
        float w = W[o * INF + i];
        s1 = fmaf(w, a1[o], s1);
        s2 = fmaf(w, a2[o], s2);
    }
    wa1[i] = s1;
    wa2[i] = s2;
}

// ---------------------------------------------------------------------------
// Kernel 2: a1v[r] = emb1[r,:]·wa1, a2v[r] = emb1[r,:]·wa2  for r in [0,B*N)
// via v_wmma_f32_16x16x32_bf16. K-loop unrolled 4x so 4 A/B load pairs are in
// flight ahead of 4 back-to-back WMMAs (bandwidth-bound on emb1).
// ---------------------------------------------------------------------------
__global__ void gat_av_wmma(const float* __restrict__ emb,
                            const float* __restrict__ wa1,
                            const float* __restrict__ wa2,
                            float* __restrict__ a1v, float* __restrict__ a2v) {
    const int lane    = threadIdx.x & 31;
    const int wave    = threadIdx.x >> 5;
    const int rowBase = blockIdx.x * 128 + wave * 16;
    const int m       = lane & 15;   // A: row within tile; B/C: column
    const int hi      = lane >> 4;   // lane-group selector

    const float* arow  = emb + (size_t)(rowBase + m) * INF;
    // B columns: 0 -> wa1, 1 -> wa2, others zero (mask trick keeps EXEC full)
    const float* bsrc  = (m == 1) ? wa2 : wa1;
    const float  bmask = (m < 2) ? 1.f : 0.f;

    v8f c = {};
    #pragma unroll 4
    for (int kk = 0; kk < INF; kk += 32) {
        // ---- A fragment: lane-group hi holds K = {hi*8..+7, 16+hi*8..+7}
        const v4f* ap4 = (const v4f*)(arow + kk + hi * 8);
        v4f x0 = ap4[0], x1 = ap4[1], x2 = ap4[4], x3 = ap4[5];
        v16bf a;
        #pragma unroll
        for (int t = 0; t < 4; ++t) {
            a[t]      = (__bf16)x0[t];
            a[4 + t]  = (__bf16)x1[t];
            a[8 + t]  = (__bf16)x2[t];
            a[12 + t] = (__bf16)x3[t];
        }
        // ---- B fragment: lane = column n, lane-group hi holds K = {hi*8..+7, 16+hi*8..+7}
        const v4f* bp4 = (const v4f*)(bsrc + kk + hi * 8);
        v4f y0 = bp4[0], y1 = bp4[1], y2 = bp4[4], y3 = bp4[5];
        v16bf bf;
        #pragma unroll
        for (int t = 0; t < 4; ++t) {
            bf[t]      = (__bf16)(y0[t] * bmask);
            bf[4 + t]  = (__bf16)(y1[t] * bmask);
            bf[8 + t]  = (__bf16)(y2[t] * bmask);
            bf[12 + t] = (__bf16)(y3[t] * bmask);
        }
        c = __builtin_amdgcn_wmma_f32_16x16x32_bf16(false, a, false, bf,
                                                    (short)0, c, false, false);
    }

    // C/D layout: lane col N = lane&15; VGPR r -> row M = r + 8*(lane>>4)
    if (m < 2) {
        float* dst = (m == 0) ? a1v : a2v;
        #pragma unroll
        for (int r = 0; r < 8; ++r)
            dst[rowBase + hi * 8 + r] = c[r];
    }
}

// ---------------------------------------------------------------------------
// Kernel 3: keep[b,j] = (j < ns_tgt[b]) && (rank of a2v[b,j] among valid < k_b)
// rank = #{j' < nt : a2v[j'] > a2v[j]} + #{j' < j : a2v[j'] == a2v[j]} (stable).
// ---------------------------------------------------------------------------
__global__ void gat_rank(const float* __restrict__ a2v,
                         const int* __restrict__ n_src, const int* __restrict__ ns_tgt,
                         int* __restrict__ keep) {
    const int b  = blockIdx.x >> 3;                       // 8 blocks per batch
    const int j  = ((blockIdx.x & 7) << 8) + threadIdx.x; // 256 j per block
    const int nt = ns_tgt[b];
    const int kb = (int)((float)n_src[b] * 0.4f);

    __shared__ float s[NN];
    for (int t = threadIdx.x; t < NN; t += blockDim.x)
        s[t] = a2v[b * NN + t];
    __syncthreads();

    int k = 0;
    if (j < nt) {
        const float v = s[j];
        int r = 0;
        #pragma unroll 4
        for (int jp = 0; jp < nt; ++jp) {
            float w = s[jp];
            r += (w > v) || ((w == v) & (jp < j));
        }
        k = (r < kb) ? 1 : 0;
    }
    keep[b * NN + j] = k;
}

// ---------------------------------------------------------------------------
// Kernel 4: out[b,i,j] = fwd&&bwd ? tanh(s_ij/8)/(0.4*n_src) : 0
//   fwd = (i<n_src) && keep[j] && (a1v[i]+a2v[j] > 0)
//   bwd = (j<n_src) && keep[i] && (a1v[j]+a2v[i] > 0)
// 134 MB of write-once output -> nontemporal B128 stores; >=60% of rows take
// the pure zero-fill fast path (k_b ~= 0.4*n_src).
// ---------------------------------------------------------------------------
__global__ void gat_out(const float* __restrict__ a1v, const float* __restrict__ a2v,
                        const int* __restrict__ keep, const int* __restrict__ n_src,
                        float* __restrict__ out) {
    const int bi = blockIdx.x;
    const int b  = bi >> 11;      // /NN
    const int i  = bi & (NN - 1);
    const int ns = n_src[b];
    float* __restrict__ orow = out + (size_t)bi * NN;

    const bool row_active = (i < ns) && keep[b * NN + i];
    if (!row_active) {
        v4f z = {0.f, 0.f, 0.f, 0.f};
        for (int j4 = threadIdx.x; j4 < NN / 4; j4 += blockDim.x)
            __builtin_nontemporal_store(z, (v4f*)orow + j4);
        return;
    }

    const float a1i     = a1v[b * NN + i];
    const float a2i     = a2v[b * NN + i];
    const float invlink = 1.0f / ((float)ns * 0.4f);
    const v4f* a1b = (const v4f*)(a1v + b * NN);
    const v4f* a2b = (const v4f*)(a2v + b * NN);
    const v4i* kb4 = (const v4i*)(keep + b * NN);

    for (int j4 = threadIdx.x; j4 < NN / 4; j4 += blockDim.x) {
        v4f a1j = a1b[j4];
        v4f a2j = a2b[j4];
        v4i kj  = kb4[j4];
        v4f r;
        #pragma unroll
        for (int t = 0; t < 4; ++t) {
            int   jj   = j4 * 4 + t;
            float s_ij = a1i + a2j[t];
            float s_ji = a1j[t] + a2i;
            bool  fwd  = (s_ij > 0.f) & (kj[t] != 0);
            bool  bwd  = (s_ji > 0.f) & (jj < ns);
            float v = 0.f;
            if (fwd & bwd) {
                // tanh(s/8) = (1 - e^{-s/4}) / (1 + e^{-s/4}), s > 0 here
                float ez = __expf(-0.25f * s_ij);
                v = invlink * (1.f - ez) / (1.f + ez);
            }
            r[t] = v;
        }
        __builtin_nontemporal_store(r, (v4f*)orow + j4);
    }
}

// ---------------------------------------------------------------------------
extern "C" void kernel_launch(void* const* d_in, const int* in_sizes, int n_in,
                              void* d_out, int out_size, void* d_ws, size_t ws_size,
                              hipStream_t stream) {
    const float* emb1   = (const float*)d_in[0];
    const float* W      = (const float*)d_in[1];
    const float* a1     = (const float*)d_in[2];
    const float* a2     = (const float*)d_in[3];
    const int*   n_src  = (const int*)d_in[4];
    const int*   ns_tgt = (const int*)d_in[5];
    float* out = (float*)d_out;

    // Workspace layout (floats): wa1[512] | wa2[512] | a1v[B*N] | a2v[B*N] | keep[B*N] (int)
    float* ws   = (float*)d_ws;
    float* wa1  = ws;
    float* wa2  = ws + INF;
    float* a1v  = ws + 2 * INF;
    float* a2v  = a1v + BATCH * NN;
    int*   keep = (int*)(a2v + BATCH * NN);

    gat_prep<<<2, 256, 0, stream>>>(W, a1, a2, wa1, wa2);
    gat_av_wmma<<<(BATCH * NN) / 128, 256, 0, stream>>>(emb1, wa1, wa2, a1v, a2v);
    gat_rank<<<BATCH * 8, 256, 0, stream>>>(a2v, n_src, ns_tgt, keep);
    gat_out<<<BATCH * NN, 256, 0, stream>>>(a1v, a2v, keep, n_src, out);
}
